// Slayer_base_16655883173895
// MI455X (gfx1250) — matrix operand order repaired and checked by reading, not compiled, not run on hardware
//
#include <hip/hip_runtime.h>
#include <hip/hip_bf16.h>

// ---------------------------------------------------------------------------
// y = A(x) @ B(weights) + bias   as a WMMA f16 GEMM with f32 accumulate.
//   A [1024 x 1152] f16 : K = i*9+f ; f<8 -> sin(grid[f]*x[b,i]) ; f=8 -> silu
//   B [1152 x  256] f16 : f<8 -> scale_sp[n]*coef[n,f] ; f=8 -> scale_base[n]
// B is stored pre-swizzled to the WMMA B-fragment layout so the GEMM wave
// does only contiguous 128-bit loads.
// ---------------------------------------------------------------------------

typedef __attribute__((ext_vector_type(16))) _Float16 v16h;
typedef __attribute__((ext_vector_type(8)))  _Float16 v8h;
typedef __attribute__((ext_vector_type(8)))  float    v8f;

#define BATCH   1024
#define IN_DIM  128
#define OUT_DIM 256
#define NUM_F   8
#define FEAT    (NUM_F + 1)          // 9 features per input dim
#define KDIM    (IN_DIM * FEAT)      // 1152
#define KBLK    32                   // WMMA K per step
#define NKB     (KDIM / KBLK)        // 36 K-steps
#define NT_N    (OUT_DIM / 16)       // 16 N tiles
#define NT_M    (BATCH / 16)         // 64 M tiles
#define NUM     (IN_DIM * OUT_DIM)   // 32768

#define A_HALVES ((size_t)BATCH * KDIM)                 // 1,179,648
#define B_HALVES ((size_t)NT_N * NKB * 32 * 16)         //   294,912

// --- Phase 1a: A[b, i*9+f] --------------------------------------------------
__global__ void __launch_bounds__(256)
build_A(const float* __restrict__ x, const float* __restrict__ grid,
        _Float16* __restrict__ A) {
  int idx = blockIdx.x * blockDim.x + threadIdx.x;   // idx = b*IN_DIM + i
  if (idx >= BATCH * IN_DIM) return;
  float xv = x[idx];
  _Float16* p = A + (size_t)idx * FEAT;              // b*KDIM + i*9 == idx*9
#pragma unroll
  for (int f = 0; f < NUM_F; ++f)
    p[f] = (_Float16)__sinf(xv * grid[f]);
  p[NUM_F] = (_Float16)(xv / (1.0f + __expf(-xv)));  // silu
}

// --- Phase 1b: B swizzled to WMMA fragment layout ---------------------------
// Fragment chunk = ((ntile*NKB + kb)*32 + lane) * 16 halves (32B contiguous).
// lane = 16*(kk/16) + (o%16), pos = kk%16  (kk = K % 32)  [contiguous k order, same as A]
__global__ void __launch_bounds__(256)
build_B(const float* __restrict__ coef, const float* __restrict__ scale_sp,
        const float* __restrict__ scale_base, _Float16* __restrict__ Bsw) {
  int t = blockIdx.x * blockDim.x + threadIdx.x;     // t over NUM*FEAT
  if (t >= NUM * FEAT) return;
  int n = t / FEAT, f = t % FEAT;
  int o = n / IN_DIM, i = n % IN_DIM;
  float v = (f < NUM_F) ? coef[(size_t)n * NUM_F + f] * scale_sp[n]
                        : scale_base[n];
  int K  = i * FEAT + f;
  int nt = o >> 4, ln = o & 15;
  int kb = K >> 5, kk = K & 31;
  int lane = ((kk >> 4) << 4) | ln;
  int pos  = kk & 15;
  Bsw[(((size_t)(nt * NKB + kb) * 32) + lane) * 16 + pos] = (_Float16)v;
}

// --- Phase 2: WMMA GEMM, one 16x16 output tile per wave ---------------------
__global__ void __launch_bounds__(128)
gemm_wmma(const _Float16* __restrict__ A, const _Float16* __restrict__ Bsw,
          const float* __restrict__ bias_w, float* __restrict__ out) {
  const int lane = threadIdx.x & 31;
  const int wave = threadIdx.x >> 5;
  const int tile = blockIdx.x * 4 + wave;            // 1024 tiles total
  const int mt = tile >> 4;                          // 0..63
  const int nt = tile & 15;                          // 0..15

  // A fragment: lane (half*16 + M) reads K chunks {0-15} / {16-31}
  const int row  = mt * 16 + (lane & 15);
  const int koff = (lane >> 4) ? 16 : 0;
  const _Float16* pa = A + (size_t)row * KDIM + koff;
  // B fragment: 32B contiguous per lane per K-step
  const _Float16* pb = Bsw + ((size_t)nt * NKB * 32 + lane) * 16;

  // C/D layout: VGPR r <-> M = r + 8*(lane>=16), N = lane%16. Seed with bias.
  float bv = bias_w[nt * 16 + (lane & 15)];
  v8f c;
#pragma unroll
  for (int r = 0; r < 8; ++r) c[r] = bv;

  for (int kb = 0; kb < NKB; ++kb) {
    v8h alo = *(const v8h*)(pa + (size_t)kb * KBLK);
    v8h ahi = *(const v8h*)(pa + (size_t)kb * KBLK + 8);
    v8h blo = *(const v8h*)(pb + (size_t)kb * 32 * 16);
    v8h bhi = *(const v8h*)(pb + (size_t)kb * 32 * 16 + 8);
    v16h a, b;
#pragma unroll
    for (int j = 0; j < 8; ++j) {
      a[j] = alo[j]; a[j + 8] = ahi[j];
      b[j] = blo[j]; b[j + 8] = bhi[j];
    }
    // (neg_a, A, neg_b, B, c_mod, C, reuse_a, reuse_b)
    c = __builtin_amdgcn_wmma_f32_16x16x32_f16(false, a, false, b,
                                               (short)0, c, false, false);
  }

  const int Nl = lane & 15;
  const int Mo = (lane >> 4) * 8;
  float* po = out + (size_t)(mt * 16 + Mo) * OUT_DIM + nt * 16 + Nl;
#pragma unroll
  for (int r = 0; r < 8; ++r) po[(size_t)r * OUT_DIM] = c[r];
}

// ---------------------------------------------------------------------------
extern "C" void kernel_launch(void* const* d_in, const int* in_sizes, int n_in,
                              void* d_out, int out_size, void* d_ws, size_t ws_size,
                              hipStream_t stream) {
  const float* x          = (const float*)d_in[0];   // [1024,128]
  const float* grid       = (const float*)d_in[1];   // [8]
  const float* coef       = (const float*)d_in[2];   // [32768,8]
  const float* scale_sp   = (const float*)d_in[3];   // [32768]
  const float* scale_base = (const float*)d_in[4];   // [32768]
  const float* bias_w     = (const float*)d_in[5];   // [1,256]
  float* out = (float*)d_out;                        // [1024,256]

  _Float16* A   = (_Float16*)d_ws;                   // 2,359,296 B
  _Float16* Bsw = A + A_HALVES;                      //   589,824 B

  build_A<<<(BATCH * IN_DIM + 255) / 256, 256, 0, stream>>>(x, grid, A);
  build_B<<<(NUM * FEAT + 255) / 256, 256, 0, stream>>>(coef, scale_sp,
                                                        scale_base, Bsw);
  gemm_wmma<<<(NT_M * NT_N) / 4, 128, 0, stream>>>(A, Bsw, bias_w, out);
}
